// Attention_9912784520067
// MI455X (gfx1250) — compile-verified, matrix operand order checked by reading, not run on hardware
//
#include <hip/hip_runtime.h>
#include <math.h>

typedef __attribute__((ext_vector_type(16))) _Float16 v16h;
typedef __attribute__((ext_vector_type(8)))  _Float16 v8h;
typedef __attribute__((ext_vector_type(4)))  _Float16 v4h;
typedef __attribute__((ext_vector_type(8)))  float    v8f;
typedef __attribute__((ext_vector_type(4)))  float    f4;

#define S_LEN 8192
#define DH    256
#define BM    128   // query rows per block (8 waves x 16)
#define WM    16    // query rows per wave
#define BK    32    // keys per main-loop iteration
#define NW    8     // waves per block

// ---- CDNA5 async copy: global -> LDS, 16 bytes per lane, tracked by ASYNCcnt ----
__device__ __forceinline__ void async_b128(unsigned lds_off, const void* gaddr) {
    asm volatile("global_load_async_to_lds_b128 %0, %1, off"
                 :: "v"(lds_off), "v"((unsigned long long)(__UINTPTR_TYPE__)gaddr)
                 : "memory");
}
__device__ __forceinline__ void wait_async0() {
    asm volatile("s_wait_asynccnt 0" ::: "memory");
}

// ---- One-time pre-pass: K (f32 row-major) -> Kh (f16 row-major) ----
__global__ void convert_k_f16(const float* __restrict__ K, _Float16* __restrict__ Kh) {
    size_t f = (size_t)blockIdx.x * blockDim.x + threadIdx.x;  // float4 index
    f4 v = *(const f4*)(K + f * 4);
    v4h h;
    h[0] = (_Float16)v[0]; h[1] = (_Float16)v[1];
    h[2] = (_Float16)v[2]; h[3] = (_Float16)v[3];
    *(v4h*)(Kh + f * 4) = h;
}

// ---- One-time pre-pass: V (f32 [S][D]) -> Vt (f16 [D][S], transposed) ----
__global__ void transpose_v_f16(const float* __restrict__ V, _Float16* __restrict__ Vt) {
    size_t f = (size_t)blockIdx.x * blockDim.x + threadIdx.x;  // float4 index
    size_t row = f >> 6;             // 0..S-1 (64 float4 per row)
    int    c4  = (int)(f & 63) << 2; // 0..252
    f4 v = *(const f4*)(V + f * 4);
    Vt[(size_t)(c4 + 0) * S_LEN + row] = (_Float16)v[0];
    Vt[(size_t)(c4 + 1) * S_LEN + row] = (_Float16)v[1];
    Vt[(size_t)(c4 + 2) * S_LEN + row] = (_Float16)v[2];
    Vt[(size_t)(c4 + 3) * S_LEN + row] = (_Float16)v[3];
}

__global__ __launch_bounds__(256)
void fa_fwd_wmma(const float* __restrict__ Q, const _Float16* __restrict__ Kh,
                 const _Float16* __restrict__ Vt, float* __restrict__ O)
{
    __shared__ _Float16 KlS[2][BK * DH];   // 2 x 16 KB, K tile row-major f16
    __shared__ _Float16 VtS[2][DH * BK];   // 2 x 16 KB, V tile transposed f16
    __shared__ _Float16 Pl[NW * WM * BK];  // 8 KB, per-wave P staging

    const int tid    = threadIdx.x;
    const int wave   = tid >> 5;
    const int lane   = tid & 31;
    const int lane16 = lane & 15;
    const int half   = lane >> 4;

    const int   qBase = blockIdx.x * BM + wave * WM;
    const float scale = 0.011048543456039806f; // 1/sqrt(8192): ref scales by sqrt(seq len)

    // ---- Load Q fragment once: 16x256 in f16 A-layout, 8 K-steps of 32 ----
    v16h qf[8];
    #pragma unroll
    for (int s = 0; s < 8; ++s) {
        const float* qp = Q + (size_t)(qBase + lane16) * DH + s * 32 + half * 8;
        f4 a0 = *(const f4*)(qp);
        f4 a1 = *(const f4*)(qp + 4);
        f4 b0 = *(const f4*)(qp + 16);
        f4 b1 = *(const f4*)(qp + 20);
        v16h q;
        #pragma unroll
        for (int i = 0; i < 4; ++i) {
            q[i]      = (_Float16)(a0[i] * scale);
            q[i + 4]  = (_Float16)(a1[i] * scale);
            q[i + 8]  = (_Float16)(b0[i] * scale);
            q[i + 12] = (_Float16)(b1[i] * scale);
        }
        qf[s] = q;
    }

    // ---- Running state (C-layout: VGPR v holds row v + 8*half) ----
    v8f acc[16];
    #pragma unroll
    for (int n = 0; n < 16; ++n) acc[n] = (v8f){0,0,0,0,0,0,0,0};
    v8f mrow = (v8f){-1e30f,-1e30f,-1e30f,-1e30f,-1e30f,-1e30f,-1e30f,-1e30f};
    v8f lrow = (v8f){0,0,0,0,0,0,0,0};

    // ---- Async tile stage: pure f16 byte copy, 8 x b128 per thread ----
    auto issue_tile = [&](int kb, int buf) {
        // K tile: contiguous 16 KB (rows kb..kb+31 of Kh)
        unsigned    klds = (unsigned)(__UINTPTR_TYPE__)&KlS[buf][0];
        const char* kgl  = (const char*)(Kh + (size_t)kb * DH);
        #pragma unroll
        for (int i = 0; i < 4; ++i)
            async_b128(klds + tid * 16 + i * 4096, kgl + tid * 16 + i * 4096);
        // Vt tile: 256 rows (d = tid) x 64 B, global row stride 16 KB
        unsigned    vlds = (unsigned)(__UINTPTR_TYPE__)&VtS[buf][0] + tid * 64;
        const char* vgl  = (const char*)(Vt + (size_t)tid * S_LEN + kb);
        #pragma unroll
        for (int j = 0; j < 4; ++j)
            async_b128(vlds + j * 16, vgl + j * 16);
    };

    issue_tile(0, 0);

    for (int kb = 0, it = 0; kb < S_LEN; kb += BK, ++it) {
        const int cur = it & 1;
        wait_async0();       // my async loads for 'cur' are in LDS
        __syncthreads();     // everyone's are; everyone done reading buffer cur^1
        if (kb + BK < S_LEN) issue_tile(kb + BK, cur ^ 1);  // prefetch next tile

        const _Float16* Kl = &KlS[cur][0];
        const _Float16* Vl = &VtS[cur][0];

        // ---- S = Q * K^T : 2 N-tiles x 8 K-steps ----
        v8f st[2];
        #pragma unroll
        for (int t = 0; t < 2; ++t) {
            v8f c = (v8f){0,0,0,0,0,0,0,0};
            #pragma unroll
            for (int s = 0; s < 8; ++s) {
                const _Float16* bp = &Kl[(t * 16 + lane16) * DH + s * 32 + half * 8];
                v8h lo = *(const v8h*)bp;
                v8h hi = *(const v8h*)(bp + 16);
                v16h b;
                #pragma unroll
                for (int i = 0; i < 8; ++i) { b[i] = lo[i]; b[i + 8] = hi[i]; }
                c = __builtin_amdgcn_wmma_f32_16x16x32_f16(false, qf[s], false, b,
                                                           (short)0, c, false, false);
            }
            st[t] = c;
        }

        // ---- Online softmax update (rows live in 16-lane groups) ----
        v8f mnew, alpha, rs;
        #pragma unroll
        for (int v = 0; v < 8; ++v) {
            float x = fmaxf(st[0][v], st[1][v]);
            #pragma unroll
            for (int off = 1; off < 16; off <<= 1)
                x = fmaxf(x, __shfl_xor(x, off, 16));
            float mn = fmaxf(mrow[v], x);
            mnew[v]  = mn;
            alpha[v] = __expf(mrow[v] - mn);
        }
        #pragma unroll
        for (int v = 0; v < 8; ++v) {
            float p0 = __expf(st[0][v] - mnew[v]);
            float p1 = __expf(st[1][v] - mnew[v]);
            st[0][v] = p0; st[1][v] = p1;
            float s2 = p0 + p1;
            #pragma unroll
            for (int off = 1; off < 16; off <<= 1)
                s2 += __shfl_xor(s2, off, 16);
            rs[v] = s2;
        }
        #pragma unroll
        for (int v = 0; v < 8; ++v) {
            lrow[v] = lrow[v] * alpha[v] + rs[v];
            mrow[v] = mnew[v];
        }
        #pragma unroll
        for (int n = 0; n < 16; ++n) {
            #pragma unroll
            for (int v = 0; v < 8; ++v) acc[n][v] *= alpha[v];
        }

        // ---- C-layout P -> LDS -> A-layout P fragment (wave-private patch) ----
        _Float16* pw = &Pl[wave * WM * BK];
        #pragma unroll
        for (int t = 0; t < 2; ++t) {
            #pragma unroll
            for (int v = 0; v < 8; ++v)
                pw[(v + 8 * half) * BK + t * 16 + lane16] = (_Float16)st[t][v];
        }
        asm volatile("s_wait_dscnt 0" ::: "memory"); // wave-private RAW through LDS
        const _Float16* pr = pw + lane16 * BK + half * 8;
        v8h plo = *(const v8h*)pr;
        v8h phi = *(const v8h*)(pr + 16);
        v16h pa;
        #pragma unroll
        for (int i = 0; i < 8; ++i) { pa[i] = plo[i]; pa[i + 8] = phi[i]; }

        // ---- O += P * V : 16 N-tiles, K=32 ----
        #pragma unroll
        for (int n = 0; n < 16; ++n) {
            const _Float16* vp = &Vl[(n * 16 + lane16) * BK + half * 8];
            v8h vlo = *(const v8h*)vp;
            v8h vhi = *(const v8h*)(vp + 16);
            v16h vb;
            #pragma unroll
            for (int i = 0; i < 8; ++i) { vb[i] = vlo[i]; vb[i + 8] = vhi[i]; }
            acc[n] = __builtin_amdgcn_wmma_f32_16x16x32_f16(false, pa, false, vb,
                                                            (short)0, acc[n], false, false);
        }
    }

    // ---- Epilogue: O = acc / l, coalesced f32 stores ----
    #pragma unroll
    for (int v = 0; v < 8; ++v) {
        float inv = 1.0f / lrow[v];
        size_t row = (size_t)(qBase + v + 8 * half);
        #pragma unroll
        for (int n = 0; n < 16; ++n)
            O[row * DH + n * 16 + lane16] = acc[n][v] * inv;
    }
}

extern "C" void kernel_launch(void* const* d_in, const int* in_sizes, int n_in,
                              void* d_out, int out_size, void* d_ws, size_t ws_size,
                              hipStream_t stream) {
    (void)in_sizes; (void)n_in; (void)out_size; (void)ws_size;
    const float* Q = (const float*)d_in[0];
    const float* K = (const float*)d_in[1];
    const float* V = (const float*)d_in[2];
    float* O = (float*)d_out;

    _Float16* Kh = (_Float16*)d_ws;                          // 4 MB
    _Float16* Vt = (_Float16*)((char*)d_ws + (size_t)S_LEN * DH * sizeof(_Float16)); // 4 MB

    const int nf4 = S_LEN * DH / 4;  // float4 count per matrix
    convert_k_f16  <<<nf4 / 256, 256, 0, stream>>>(K, Kh);
    transpose_v_f16<<<nf4 / 256, 256, 0, stream>>>(V, Vt);
    fa_fwd_wmma    <<<S_LEN / BM, 256, 0, stream>>>(Q, Kh, Vt, O);
}
